// TransformerBlock_23519240913427
// MI455X (gfx1250) — matrix operand order repaired and checked by reading, not compiled, not run on hardware
//
#include <hip/hip_runtime.h>
#include <stdint.h>

typedef __bf16 bf16;
typedef __attribute__((ext_vector_type(16))) __bf16 bf16x16;
typedef __attribute__((ext_vector_type(8)))  __bf16 bf16x8;
typedef __attribute__((ext_vector_type(8)))  float  f32x8;

#define NPTS 16384   // N per batch (reference constant)
#define PPW  8       // points per workgroup in k3

// ---------------- WMMA helpers (gfx1250, wave32) ----------------
// A 16x32 bf16 fragment from row-major source (stride ld elements).
// Lane l: row m = l&15, half h = l>>4; elems 0..7 = K[k0+8h .. +7],
// elems 8..15 = K[k0+16+8h .. +7]  (ISA 7.12.2, 16-bit A 16x32 table)
__device__ __forceinline__ bf16x16 load_a_frag(const bf16* base, int ld, int k0) {
  int lane = threadIdx.x & 31;
  int m = lane & 15, h = lane >> 4;
  const bf16* p = base + m * ld + k0 + 8 * h;
  bf16x8 lo = *(const bf16x8*)(p);
  bf16x8 hi = *(const bf16x8*)(p + 16);
  return __builtin_shufflevector(lo, hi, 0,1,2,3,4,5,6,7,8,9,10,11,12,13,14,15);
}

// B 32x16 bf16 fragment from TRANSPOSED weight Wt[n][k] (row stride ld = K-dim).
// Lane l: col n = n0 + (l&15), half h = l>>4; elems 0..7 = Wt[n][k0+8h .. +7], elems 8..15 = Wt[n][k0+16+8h .. +7]
__device__ __forceinline__ bf16x16 load_b_frag(const bf16* wt, int ld, int n0, int k0) {
  int lane = threadIdx.x & 31;
  int n = n0 + (lane & 15), h = lane >> 4;
  const bf16* p = wt + n * ld + k0 + 8 * h;
  bf16x8 lo = *(const bf16x8*)(p);
  bf16x8 hi = *(const bf16x8*)(p + 16);
  return __builtin_shufflevector(lo, hi, 0,1,2,3,4,5,6,7,8,9,10,11,12,13,14,15);
}

__device__ __forceinline__ f32x8 wmma_bf16(bf16x16 a, bf16x16 b, f32x8 c) {
  return __builtin_amdgcn_wmma_f32_16x16x32_bf16(false, a, false, b, (short)0, c, false, false);
}

// ---------------- k0: weight transpose + f32->bf16 ----------------
// W row-major [kd][128] -> Wt row-major [128][kd], Wt[n][k] = W[k][n]
__global__ void k0_conv(const float* __restrict__ W, bf16* __restrict__ Wt, int kd) {
  int i = blockIdx.x * 256 + threadIdx.x;
  if (i >= kd * 128) return;
  int n = i / kd, k = i % kd;
  Wt[i] = (bf16)W[k * 128 + n];
}

// ---------------- k1: f = feat@fc1+b (bf16 out), sc -> d_out, q = f@wq ----
__global__ __launch_bounds__(256) void k1_front(
    const float* __restrict__ feature, const float* __restrict__ fc1_b,
    const float* __restrict__ sc_b,
    const bf16* __restrict__ wt_fc1, const bf16* __restrict__ wt_sc,
    const bf16* __restrict__ wt_wq,
    bf16* __restrict__ f_bf16, float* __restrict__ q_out, float* __restrict__ out)
{
  __shared__ __align__(16) bf16 sFeat[64 * 64];
  __shared__ __align__(16) bf16 sF[64 * 128];
  const int tid = threadIdx.x;
  const long row0 = (long)blockIdx.x * 64;

  const float* fsrc = feature + row0 * 64;
  #pragma unroll
  for (int i = tid; i < 64 * 64; i += 256) sFeat[i] = (bf16)fsrc[i];
  __syncthreads();

  const int w = tid >> 5, lane = tid & 31;
  const int mt = w & 3, ng = w >> 2;      // 4 M-tiles x 2 N-groups
  const int m0 = mt * 16;
  const int colL = lane & 15, h = lane >> 4;

  bf16x16 a0 = load_a_frag(sFeat + m0 * 64, 64, 0);
  bf16x16 a1 = load_a_frag(sFeat + m0 * 64, 64, 32);

  #pragma unroll
  for (int t = 0; t < 4; ++t) {
    const int n0 = (ng * 4 + t) * 16;
    f32x8 c = {};
    c = wmma_bf16(a0, load_b_frag(wt_fc1, 64, n0, 0),  c);
    c = wmma_bf16(a1, load_b_frag(wt_fc1, 64, n0, 32), c);
    const float b1 = fc1_b[n0 + colL];
    #pragma unroll
    for (int r = 0; r < 8; ++r)
      sF[(m0 + r + 8 * h) * 128 + n0 + colL] = (bf16)(c[r] + b1);
    f32x8 s = {};
    s = wmma_bf16(a0, load_b_frag(wt_sc, 64, n0, 0),  s);
    s = wmma_bf16(a1, load_b_frag(wt_sc, 64, n0, 32), s);
    const float b2 = sc_b[n0 + colL];
    #pragma unroll
    for (int r = 0; r < 8; ++r)
      out[(row0 + m0 + r + 8 * h) * 128 + n0 + colL] = s[r] + b2;
  }
  __syncthreads();

  {
    uint32_t* dst = (uint32_t*)(f_bf16 + row0 * 128);
    const uint32_t* src = (const uint32_t*)sF;
    #pragma unroll
    for (int i = tid; i < 64 * 128 / 2; i += 256) dst[i] = src[i];
  }
  #pragma unroll
  for (int t = 0; t < 4; ++t) {
    const int n0 = (ng * 4 + t) * 16;
    f32x8 c = {};
    #pragma unroll
    for (int kb = 0; kb < 4; ++kb)
      c = wmma_bf16(load_a_frag(sF + m0 * 128, 128, kb * 32),
                    load_b_frag(wt_wq, 128, n0, kb * 32), c);
    #pragma unroll
    for (int r = 0; r < 8; ++r)
      q_out[(row0 + m0 + r + 8 * h) * 128 + n0 + colL] = c[r];
  }
}

// ---------------- k3: fused attention, PPW points per workgroup ----------
// Weight B-fragments (each wave's 16-col slice of all 5 matrices) are hoisted
// into registers once per workgroup: 20 frags x 8 VGPR = 160 VGPRs, leaving
// the steady-state loop as ds_load_b128 (A) + v_wmma + VALU epilogue.
__global__ __launch_bounds__(256) void k3_attn(
    const float* __restrict__ rel_xyz, const int* __restrict__ knn_idx,
    const float* __restrict__ fd_w1, const float* __restrict__ fd_b1,
    const float* __restrict__ fd_b2, const float* __restrict__ fg_b1,
    const float* __restrict__ fg_b2,
    const bf16* __restrict__ wt_fd2, const bf16* __restrict__ wt_wk,
    const bf16* __restrict__ wt_wv,  const bf16* __restrict__ wt_fg1,
    const bf16* __restrict__ wt_fg2,
    const bf16* __restrict__ f_bf16, const float* __restrict__ q,
    bf16* __restrict__ y_out)
{
  __shared__ __align__(16) bf16 sA[16 * 128];
  __shared__ __align__(16) bf16 sB[16 * 128];
  __shared__ float sRel[16 * 3];
  __shared__ int sIdx[16];

  const int tid = threadIdx.x;
  const int w = tid >> 5, lane = tid & 31;
  const int col = w * 16 + (lane & 15);
  const int h = lane >> 4;
  const int n0 = w * 16;
  const int gk = tid >> 4, gcp = tid & 15;    // gather role: neighbor, col-chunk
  const long p0 = (long)blockIdx.x * PPW;
  const long batch_base = (p0 / NPTS) * NPTS; // PPW divides N so block stays in one batch

  // hoisted per-wave weight slices (registers)
  bf16x16 Bk[4], Bv[4], Bp[4], Bg1[4], Bg2[4];
  #pragma unroll
  for (int kb = 0; kb < 4; ++kb) {
    Bk[kb]  = load_b_frag(wt_wk,  128, n0, kb * 32);
    Bv[kb]  = load_b_frag(wt_wv,  128, n0, kb * 32);
    Bp[kb]  = load_b_frag(wt_fd2, 128, n0, kb * 32);
    Bg1[kb] = load_b_frag(wt_fg1, 128, n0, kb * 32);
    Bg2[kb] = load_b_frag(wt_fg2, 128, n0, kb * 32);
  }
  // hoisted per-wave scalars
  const float fw0 = fd_w1[col], fw1 = fd_w1[128 + col], fw2 = fd_w1[256 + col];
  const float fb1 = fd_b1[col], pb = fd_b2[col];
  const float gb1 = fg_b1[col], gb2 = fg_b2[col];
  const float scale = 0.08838834764831843f; // 1/sqrt(128)

  #pragma unroll 1
  for (int pi = 0; pi < PPW; ++pi) {
    const long p = p0 + pi;
    if (pi + 1 < PPW) {                     // prefetch next point's metadata
      __builtin_prefetch(knn_idx + (p + 1) * 16, 0, 0);
      __builtin_prefetch(rel_xyz + (p + 1) * 48, 0, 0);
    }
    if (tid < 16) sIdx[tid] = knn_idx[p * 16 + tid];
    if (tid < 48) sRel[tid] = rel_xyz[p * 48 + tid];
    __syncthreads();   // also fences previous iteration's sA/sB reads

    // gather 16 neighbor feature rows (bf16) into sA
    {
      const long row = batch_base + (long)sIdx[gk];
      *(bf16x8*)(sA + gk * 128 + gcp * 8) =
          *(const bf16x8*)(f_bf16 + row * 128 + gcp * 8);
    }
    // pos-enc layer 1 (K=3, VALU) in C layout -> sB as bf16
    #pragma unroll
    for (int r = 0; r < 8; ++r) {
      const int m = r + 8 * h;
      float v = sRel[m * 3 + 0] * fw0 + sRel[m * 3 + 1] * fw1 +
                sRel[m * 3 + 2] * fw2 + fb1;
      sB[m * 128 + col] = (bf16)fmaxf(v, 0.f);
    }
    __syncthreads();

    // k = F@wk, v = F@wv, pos = H1@fd_w2 + b2
    f32x8 acck = {}, accv = {}, accp = {};
    #pragma unroll
    for (int kb = 0; kb < 4; ++kb) {
      bf16x16 aK = load_a_frag(sA, 128, kb * 32);
      bf16x16 aP = load_a_frag(sB, 128, kb * 32);
      acck = wmma_bf16(aK, Bk[kb], acck);
      accv = wmma_bf16(aK, Bv[kb], accv);
      accp = wmma_bf16(aP, Bp[kb], accp);
    }
    const float qv = q[p * 128 + col];
    __syncthreads();                        // all waves done reading sA/sB

    // s = q - k + pos -> sA (bf16); keep pos in accp
    #pragma unroll
    for (int r = 0; r < 8; ++r) {
      const float pos = accp[r] + pb;
      accp[r] = pos;
      sA[(r + 8 * h) * 128 + col] = (bf16)(qv - acck[r] + pos);
    }
    __syncthreads();

    // gamma MLP layer 1: relu(S@fg_w1 + b1) -> sB
    f32x8 acch = {};
    #pragma unroll
    for (int kb = 0; kb < 4; ++kb)
      acch = wmma_bf16(load_a_frag(sA, 128, kb * 32), Bg1[kb], acch);
    #pragma unroll
    for (int r = 0; r < 8; ++r)
      sB[(r + 8 * h) * 128 + col] = (bf16)fmaxf(acch[r] + gb1, 0.f);
    __syncthreads();

    // gamma MLP layer 2: A = H@fg_w2 + b2
    f32x8 acca = {};
    #pragma unroll
    for (int kb = 0; kb < 4; ++kb)
      acca = wmma_bf16(load_a_frag(sB, 128, kb * 32), Bg2[kb], acca);

    // softmax over the 16 neighbors (rows), per channel.
    float l[8], mx = -1e30f;
    #pragma unroll
    for (int r = 0; r < 8; ++r) { l[r] = (acca[r] + gb2) * scale; mx = fmaxf(mx, l[r]); }
    mx = fmaxf(mx, __shfl_xor(mx, 16));
    float e[8], se = 0.f;
    #pragma unroll
    for (int r = 0; r < 8; ++r) { e[r] = __expf(l[r] - mx); se += e[r]; }
    se += __shfl_xor(se, 16);
    const float inv = 1.f / se;

    float y = 0.f;
    #pragma unroll
    for (int r = 0; r < 8; ++r) y += e[r] * inv * (accv[r] + accp[r]);
    y += __shfl_xor(y, 16);
    if (h == 0) y_out[p * 128 + col] = (bf16)y;
  }
}

// ---------------- k4: out += y @ fc2 + fc2_b ----------------
__global__ __launch_bounds__(256) void k4_fc2(
    const bf16* __restrict__ y_bf16, const bf16* __restrict__ wt_fc2,
    const float* __restrict__ fc2_b, float* __restrict__ out)
{
  const long row0 = (long)blockIdx.x * 64;
  const int tid = threadIdx.x, w = tid >> 5, lane = tid & 31;
  const int mt = w & 3, ng = w >> 2;
  const int m0 = mt * 16;
  const int colL = lane & 15, h = lane >> 4;
  const bf16* abase = y_bf16 + (row0 + m0) * 128;
  #pragma unroll
  for (int t = 0; t < 4; ++t) {
    const int n0 = (ng * 4 + t) * 16;
    f32x8 c = {};
    #pragma unroll
    for (int kb = 0; kb < 4; ++kb)
      c = wmma_bf16(load_a_frag(abase, 128, kb * 32),
                    load_b_frag(wt_fc2, 128, n0, kb * 32), c);
    const float b = fc2_b[n0 + colL];
    #pragma unroll
    for (int r = 0; r < 8; ++r) {
      const long o = (row0 + m0 + r + 8 * h) * 128 + n0 + colL;
      out[o] = out[o] + c[r] + b;
    }
  }
}

// ---------------- host launcher ----------------
extern "C" void kernel_launch(void* const* d_in, const int* in_sizes, int n_in,
                              void* d_out, int out_size, void* d_ws, size_t ws_size,
                              hipStream_t stream) {
  const float* feature = (const float*)d_in[1];
  const float* rel     = (const float*)d_in[2];
  const int*   knn     = (const int*)d_in[3];
  const float* fd_w1   = (const float*)d_in[4];
  const float* fd_b1   = (const float*)d_in[5];
  const float* fd_w2   = (const float*)d_in[6];
  const float* fd_b2   = (const float*)d_in[7];
  const float* fc1_w   = (const float*)d_in[8];
  const float* fc1_b   = (const float*)d_in[9];
  const float* wq      = (const float*)d_in[10];
  const float* wk      = (const float*)d_in[11];
  const float* wv      = (const float*)d_in[12];
  const float* fg_w1   = (const float*)d_in[13];
  const float* fg_b1   = (const float*)d_in[14];
  const float* fg_w2   = (const float*)d_in[15];
  const float* fg_b2   = (const float*)d_in[16];
  const float* fc2_w   = (const float*)d_in[17];
  const float* fc2_b   = (const float*)d_in[18];
  const float* sc_w    = (const float*)d_in[19];
  const float* sc_b    = (const float*)d_in[20];
  float* out = (float*)d_out;

  const long rows = (long)in_sizes[1] / 64;   // B*N

  char* ws = (char*)d_ws;
  bf16* wt_fc1 = (bf16*)(ws + 0);             // 128x64
  bf16* wt_sc  = (bf16*)(ws + 16384);         // 128x64
  bf16* wt_wq  = (bf16*)(ws + 32768);
  bf16* wt_wk  = (bf16*)(ws + 65536);
  bf16* wt_wv  = (bf16*)(ws + 98304);
  bf16* wt_fg1 = (bf16*)(ws + 131072);
  bf16* wt_fg2 = (bf16*)(ws + 163840);
  bf16* wt_fd2 = (bf16*)(ws + 196608);
  bf16* wt_fc2 = (bf16*)(ws + 229376);
  bf16*  f_bf  = (bf16*)(ws + 262144);                       // rows*128 bf16
  float* q_buf = (float*)(ws + 262144 + rows * 128 * 2);     // rows*128 f32
  bf16*  y_buf = (bf16*)(ws + 262144 + rows * 128 * 6);      // rows*128 bf16

  const int g64 = (64 * 128 + 255) / 256;
  const int g128 = (128 * 128 + 255) / 256;
  k0_conv<<<g64,  256, 0, stream>>>(fc1_w, wt_fc1, 64);
  k0_conv<<<g64,  256, 0, stream>>>(sc_w,  wt_sc,  64);
  k0_conv<<<g128, 256, 0, stream>>>(wq,    wt_wq,  128);
  k0_conv<<<g128, 256, 0, stream>>>(wk,    wt_wk,  128);
  k0_conv<<<g128, 256, 0, stream>>>(wv,    wt_wv,  128);
  k0_conv<<<g128, 256, 0, stream>>>(fg_w1, wt_fg1, 128);
  k0_conv<<<g128, 256, 0, stream>>>(fg_w2, wt_fg2, 128);
  k0_conv<<<g128, 256, 0, stream>>>(fd_w2, wt_fd2, 128);
  k0_conv<<<g128, 256, 0, stream>>>(fc2_w, wt_fc2, 128);

  k1_front<<<(int)(rows / 64), 256, 0, stream>>>(
      feature, fc1_b, sc_b, wt_fc1, wt_sc, wt_wq, f_bf, q_buf, out);

  k3_attn<<<(int)(rows / PPW), 256, 0, stream>>>(
      rel, knn, fd_w1, fd_b1, fd_b2, fg_b1, fg_b2,
      wt_fd2, wt_wk, wt_wv, wt_fg1, wt_fg2, f_bf, q_buf, y_buf);

  k4_fc2<<<(int)(rows / 64), 256, 0, stream>>>(y_buf, wt_fc2, fc2_b, out);
}
